// VanillaLLMVM_62380105007360
// MI455X (gfx1250) — matrix operand order repaired and checked by reading, not compiled, not run on hardware
//
#include <hip/hip_runtime.h>

// ---------------------------------------------------------------------------
// CDNA5 (gfx1250) transformer layer: LN -> QKV -> flash-attn -> O+resid ->
// LN -> FFN_in -> (u@sw_w, u@sw_v, SwiGLU) -> FFN_out + resid.
// GEMMs: v_wmma_f32_16x16x32_bf16, double-buffered LDS, async global->LDS
// copies (ASYNCcnt) when the toolchain exposes them.
// ---------------------------------------------------------------------------

typedef unsigned short bf16_t;
typedef __attribute__((ext_vector_type(16))) __bf16 v16bf;
typedef __attribute__((ext_vector_type(8)))  float  v8f;

union FragAB { v16bf v; uint4 u[2]; };

// Async global->LDS copy path (gfx1250): guarded so the file compiles on
// toolchains that don't declare the builtins.
#if defined(__has_builtin)
#  if __has_builtin(__builtin_amdgcn_global_load_async_to_lds_b128) && \
      __has_builtin(__builtin_amdgcn_s_wait_asynccnt)
#    define GEMM_ASYNC 1
#  endif
#endif
#ifndef GEMM_ASYNC
#  define GEMM_ASYNC 0
#endif

#if GEMM_ASYNC
// Builtin signature (from hipcc diagnostic): param type is
// 'vector_size(16) int __device__ *' i.e. v4i in address-space 1 (global);
// LDS side is the address-space 3 analogue.
typedef int v4i_gen __attribute__((vector_size(16)));
typedef __attribute__((address_space(1))) v4i_gen as1_v4i;
typedef __attribute__((address_space(3))) v4i_gen as3_v4i;
// Flat LDS addresses carry the byte offset in the low 32 bits (aperture in
// the high bits), so trunc(flat) is the LDS offset the async unit wants.
__device__ __forceinline__ as3_v4i* to_lds(void* p) {
  return (as3_v4i*)(unsigned)(size_t)p;
}
__device__ __forceinline__ as1_v4i* to_glb(const void* p) {
  return (as1_v4i*)(size_t)p;
}
#endif

__device__ __forceinline__ bf16_t f2bf(float x) {
  union { float f; unsigned u; } v; v.f = x;
  unsigned r = v.u + 0x7FFFu + ((v.u >> 16) & 1u);   // round-to-nearest-even
  return (bf16_t)(r >> 16);
}

// ---------------------------------------------------------------- cast kernel
__global__ void cast_f32_to_bf16(const float* __restrict__ src,
                                 bf16_t* __restrict__ dst, int n) {
  int i = blockIdx.x * blockDim.x + threadIdx.x;
  int stride = gridDim.x * blockDim.x;
  for (; i < n; i += stride) dst[i] = f2bf(src[i]);
}

// ---------------------------------------------------------------- layernorm
// One wave32 per 1024-element row; 8 rows per 256-thread block.
__global__ void __launch_bounds__(256)
layernorm_bf16(const float* __restrict__ x, const float* __restrict__ g,
               const float* __restrict__ b, bf16_t* __restrict__ out, int rows) {
  const int D = 1024;
  int row = blockIdx.x * 8 + (threadIdx.x >> 5);
  if (row >= rows) return;
  int lane = threadIdx.x & 31;
  const float* xr = x + (size_t)row * D;
  float v[32];
  float s = 0.f;
#pragma unroll
  for (int i = 0; i < 32; i++) { v[i] = xr[lane + i * 32]; s += v[i]; }
#pragma unroll
  for (int o = 16; o > 0; o >>= 1) s += __shfl_xor(s, o, 32);
  float mu = s * (1.f / 1024.f);
  float q = 0.f;
#pragma unroll
  for (int i = 0; i < 32; i++) { float d = v[i] - mu; q += d * d; }
#pragma unroll
  for (int o = 16; o > 0; o >>= 1) q += __shfl_xor(q, o, 32);
  float rstd = rsqrtf(q * (1.f / 1024.f) + 1e-5f);
  bf16_t* orow = out + (size_t)row * D;
#pragma unroll
  for (int i = 0; i < 32; i++) {
    int c = lane + i * 32;
    orow[c] = f2bf((v[i] - mu) * rstd * g[c] + b[c]);
  }
}

// ---------------------------------------------------------------- WMMA GEMM
// C[M,N] = A[M,K](bf16) x W[N,K]^T(bf16), f32 accumulate.
// 256 thr = 8 waves, tile 128x128, BK=32, ping-pong LDS. Wave tile 64x32.
// EPI: 0 = f32 store (+bias/+resid), 1 = bf16 store (+bias), 2 = SwiGLU:
//      out_bf16 = silu(aux) * acc.
template <int EPI>
__global__ void __launch_bounds__(256)
gemm_wmma_bf16(const bf16_t* __restrict__ A, const bf16_t* __restrict__ W,
               const float* __restrict__ bias, const float* __restrict__ resid,
               const float* __restrict__ aux, float* __restrict__ outF,
               bf16_t* __restrict__ outB, int M, int N, int K) {
  constexpr int LDT = 40;                 // 32 + 8 pad (16B-aligned rows)
  __shared__ bf16_t As[2][128 * LDT];
  __shared__ bf16_t Bs[2][128 * LDT];

  const int tid = threadIdx.x;
  const int m0 = blockIdx.y * 128;
  const int n0 = blockIdx.x * 128;
  const int wave = tid >> 5, lane = tid & 31;
  const int lh = lane >> 4, lr = lane & 15;
  const int waveM = wave >> 2;            // 0..1 -> 64 rows
  const int waveN = wave & 3;             // 0..3 -> 32 cols

  const int ldRow = tid >> 1;             // 0..127
  const int ldCol = (tid & 1) * 16;       // 0 or 16
  const size_t aRowOff = (size_t)(m0 + ldRow) * K + ldCol;
  const size_t wRowOff = (size_t)(n0 + ldRow) * K + ldCol;
  const int ldsOff = ldRow * LDT + ldCol;

  v8f acc[4][2] = {};

#if GEMM_ASYNC
  // -------- async (ASYNCcnt) double-buffered pipeline --------
  {
    __builtin_amdgcn_global_load_async_to_lds_b128(
        to_glb(A + aRowOff), to_lds(&As[0][ldsOff]), 0, 0);
    __builtin_amdgcn_global_load_async_to_lds_b128(
        to_glb(A + aRowOff), to_lds(&As[0][ldsOff]), 16, 0);
    __builtin_amdgcn_global_load_async_to_lds_b128(
        to_glb(W + wRowOff), to_lds(&Bs[0][ldsOff]), 0, 0);
    __builtin_amdgcn_global_load_async_to_lds_b128(
        to_glb(W + wRowOff), to_lds(&Bs[0][ldsOff]), 16, 0);
    __builtin_amdgcn_s_wait_asynccnt(0);
  }
  __syncthreads();
#else
  // -------- fallback: register-staged double-buffered pipeline --------
  {
    uint4 a0 = *(const uint4*)(A + aRowOff);
    uint4 a1 = *(const uint4*)(A + aRowOff + 8);
    uint4 b0 = *(const uint4*)(W + wRowOff);
    uint4 b1 = *(const uint4*)(W + wRowOff + 8);
    *(uint4*)(&As[0][ldsOff])     = a0;
    *(uint4*)(&As[0][ldsOff + 8]) = a1;
    *(uint4*)(&Bs[0][ldsOff])     = b0;
    *(uint4*)(&Bs[0][ldsOff + 8]) = b1;
  }
  __syncthreads();
#endif

  int cur = 0;
  for (int kt = 0; kt < K; kt += 32) {
    const bool hasNext = (kt + 32) < K;
    const int nxt = cur ^ 1;

#if GEMM_ASYNC
    if (hasNext) {                        // prefetch next tile via async DMA
      __builtin_amdgcn_global_load_async_to_lds_b128(
          to_glb(A + aRowOff + kt + 32), to_lds(&As[nxt][ldsOff]), 0, 0);
      __builtin_amdgcn_global_load_async_to_lds_b128(
          to_glb(A + aRowOff + kt + 32), to_lds(&As[nxt][ldsOff]), 16, 0);
      __builtin_amdgcn_global_load_async_to_lds_b128(
          to_glb(W + wRowOff + kt + 32), to_lds(&Bs[nxt][ldsOff]), 0, 0);
      __builtin_amdgcn_global_load_async_to_lds_b128(
          to_glb(W + wRowOff + kt + 32), to_lds(&Bs[nxt][ldsOff]), 16, 0);
    }
#else
    uint4 na0, na1, nb0, nb1;
    if (hasNext) {                        // global loads overlap compute below
      na0 = *(const uint4*)(A + aRowOff + kt + 32);
      na1 = *(const uint4*)(A + aRowOff + kt + 32 + 8);
      nb0 = *(const uint4*)(W + wRowOff + kt + 32);
      nb1 = *(const uint4*)(W + wRowOff + kt + 32 + 8);
      __builtin_prefetch(A + aRowOff + kt + 64, 0, 1);
      __builtin_prefetch(W + wRowOff + kt + 64, 0, 1);
    }
#endif

    // ---- compute current tile ----
    FragAB bf[2];
#pragma unroll
    for (int j = 0; j < 2; j++) {         // B: lane=col, K chunks {lh*8,16+lh*8}
      const int wrow = waveN * 32 + j * 16 + lr;
      bf[j].u[0] = *(const uint4*)(&Bs[cur][wrow * LDT + lh * 8]);
      bf[j].u[1] = *(const uint4*)(&Bs[cur][wrow * LDT + 16 + lh * 8]);
    }
#pragma unroll
    for (int i = 0; i < 4; i++) {         // A: lane=row, K chunks {lh*8,16+lh*8}
      const int arow = waveM * 64 + i * 16 + lr;
      FragAB af;
      af.u[0] = *(const uint4*)(&As[cur][arow * LDT + lh * 8]);
      af.u[1] = *(const uint4*)(&As[cur][arow * LDT + 16 + lh * 8]);
#pragma unroll
      for (int j = 0; j < 2; j++) {
        acc[i][j] = __builtin_amdgcn_wmma_f32_16x16x32_bf16(
            false, af.v, false, bf[j].v, (short)0, acc[i][j], false, false);
      }
    }

    if (hasNext) {
#if GEMM_ASYNC
      __builtin_amdgcn_s_wait_asynccnt(0);
      __syncthreads();                    // next buffer ready for all waves
#else
      __syncthreads();                    // all waves done reading nxt's old data
      *(uint4*)(&As[nxt][ldsOff])     = na0;
      *(uint4*)(&As[nxt][ldsOff + 8]) = na1;
      *(uint4*)(&Bs[nxt][ldsOff])     = nb0;
      *(uint4*)(&Bs[nxt][ldsOff + 8]) = nb1;
      __syncthreads();
#endif
      cur = nxt;
    }
  }

  // Epilogue. C layout: VGPR r, lane -> row r + 8*lh, col lr.
#pragma unroll
  for (int i = 0; i < 4; i++) {
#pragma unroll
    for (int j = 0; j < 2; j++) {
      const int gcol = n0 + waveN * 32 + j * 16 + lr;
      const float bv = bias ? bias[gcol] : 0.f;
#pragma unroll
      for (int r = 0; r < 8; r++) {
        const int grow = m0 + waveM * 64 + i * 16 + r + 8 * lh;
        const size_t idx = (size_t)grow * N + gcol;
        float v = acc[i][j][r] + bv;
        if (resid) v += resid[idx];
        if (EPI == 2) { float a = aux[idx]; v *= a / (1.f + __expf(-a)); }
        if (EPI == 0) outF[idx] = v;
        else          outB[idx] = f2bf(v);
      }
    }
  }
}

// ---------------------------------------------------------------- flash attn
// grid (L/64, HEADS, B), block 128 = 4 waves; wave owns 16 queries.
// K-tile row-major in LDS; V-tile transposed so PV B-frags are contiguous.
__global__ void __launch_bounds__(128)
flash_attn_wmma(const bf16_t* __restrict__ qb, const bf16_t* __restrict__ kb,
                const bf16_t* __restrict__ vb, bf16_t* __restrict__ ob) {
  const int L = 2048, DM = 1024, HD = 64;
  const float scale = 0.125f;                        // 1/sqrt(64)
  const int h = blockIdx.y, b = blockIdx.z;
  const int q0 = blockIdx.x * 64;
  const int tid = threadIdx.x;
  const int wave = tid >> 5, lane = tid & 31;
  const int lh = lane >> 4, lr = lane & 15;
  const size_t base = (size_t)b * L * DM + h * HD;

  __shared__ bf16_t Ks[64 * 72];                     // [key][d]
  __shared__ bf16_t VsT[64 * 72];                    // [d][key]
  __shared__ bf16_t Ps[4 * 16 * 72];                 // per-wave P scratch

  FragAB qf[2];                                      // Q as A-frags over d
  {
    const int qrow = q0 + wave * 16 + lr;
    const bf16_t* qp = qb + base + (size_t)qrow * DM;
#pragma unroll
    for (int d = 0; d < 2; d++) {
      qf[d].u[0] = *(const uint4*)(qp + d * 32 + lh * 8);
      qf[d].u[1] = *(const uint4*)(qp + d * 32 + 16 + lh * 8);
    }
  }

  v8f o[4] = {};
  float m_i[8], l_i[8];
#pragma unroll
  for (int r = 0; r < 8; r++) { m_i[r] = -1e30f; l_i[r] = 0.f; }

  const int ldRow = tid >> 1;                        // 0..63 (key row)
  const int ldCol = (tid & 1) * 32;                  // 0 or 32 (d)

  for (int kt = 0; kt < L; kt += 64) {
    {
      const bf16_t* kp = kb + base + (size_t)(kt + ldRow) * DM + ldCol;
      *(uint4*)(&Ks[ldRow * 72 + ldCol])      = *(const uint4*)(kp);
      *(uint4*)(&Ks[ldRow * 72 + ldCol + 8])  = *(const uint4*)(kp + 8);
      *(uint4*)(&Ks[ldRow * 72 + ldCol + 16]) = *(const uint4*)(kp + 16);
      *(uint4*)(&Ks[ldRow * 72 + ldCol + 24]) = *(const uint4*)(kp + 24);
      const bf16_t* vp = vb + base + (size_t)(kt + ldRow) * DM + ldCol;
      union { uint4 u[4]; bf16_t s[32]; } vt;
      vt.u[0] = *(const uint4*)(vp);
      vt.u[1] = *(const uint4*)(vp + 8);
      vt.u[2] = *(const uint4*)(vp + 16);
      vt.u[3] = *(const uint4*)(vp + 24);
#pragma unroll
      for (int ii = 0; ii < 32; ii++)               // transpose into LDS
        VsT[(ldCol + ii) * 72 + ldRow] = vt.s[ii];
    }
    __syncthreads();

    // S = Q K^T : 16 queries x 64 keys, K-dim 64 = 2 WMMA steps
    v8f s[4] = {};
#pragma unroll
    for (int j = 0; j < 4; j++) {
#pragma unroll
      for (int d = 0; d < 2; d++) {
        FragAB kf;
        const int krow = j * 16 + lr;
        kf.u[0] = *(const uint4*)(&Ks[krow * 72 + d * 32 + lh * 8]);
        kf.u[1] = *(const uint4*)(&Ks[krow * 72 + d * 32 + 16 + lh * 8]);
        s[j] = __builtin_amdgcn_wmma_f32_16x16x32_bf16(
            false, qf[d].v, false, kf.v, (short)0, s[j], false, false);
      }
    }

    // Online softmax: lane holds col lr of rows r+8*lh; row reductions via
    // shfl_xor within the 16-lane half (xor<16 preserves bit4).
#pragma unroll
    for (int r = 0; r < 8; r++) {
      float rm = -1e30f;
#pragma unroll
      for (int j = 0; j < 4; j++) rm = fmaxf(rm, s[j][r] * scale);
#pragma unroll
      for (int off = 1; off < 16; off <<= 1)
        rm = fmaxf(rm, __shfl_xor(rm, off, 32));
      float mnew = fmaxf(m_i[r], rm);
      float corr = __expf(m_i[r] - mnew);
      float rs = 0.f;
#pragma unroll
      for (int j = 0; j < 4; j++) {
        float p = __expf(s[j][r] * scale - mnew);
        s[j][r] = p;
        rs += p;
      }
#pragma unroll
      for (int off = 1; off < 16; off <<= 1) rs += __shfl_xor(rs, off, 32);
      l_i[r] = l_i[r] * corr + rs;
      m_i[r] = mnew;
#pragma unroll
      for (int f = 0; f < 4; f++) o[f][r] *= corr;
    }

    // P (C layout) -> per-wave LDS -> A layout (same-wave LDS is in-order)
    bf16_t* pw = &Ps[wave * 16 * 72];
#pragma unroll
    for (int j = 0; j < 4; j++)
#pragma unroll
      for (int r = 0; r < 8; r++)
        pw[(r + 8 * lh) * 72 + j * 16 + lr] = f2bf(s[j][r]);

    // O += P V  (K-dim = 64 keys = 2 WMMA steps)
#pragma unroll
    for (int kc = 0; kc < 2; kc++) {
      FragAB pf;
      pf.u[0] = *(const uint4*)(&pw[lr * 72 + kc * 32 + lh * 8]);
      pf.u[1] = *(const uint4*)(&pw[lr * 72 + kc * 32 + 16 + lh * 8]);
#pragma unroll
      for (int f = 0; f < 4; f++) {
        FragAB vf;
        const int dcol = f * 16 + lr;
        vf.u[0] = *(const uint4*)(&VsT[dcol * 72 + kc * 32 + lh * 8]);
        vf.u[1] = *(const uint4*)(&VsT[dcol * 72 + kc * 32 + 16 + lh * 8]);
        o[f] = __builtin_amdgcn_wmma_f32_16x16x32_bf16(
            false, pf.v, false, vf.v, (short)0, o[f], false, false);
      }
    }
    __syncthreads();
  }

  const int qrow = q0 + wave * 16;
#pragma unroll
  for (int r = 0; r < 8; r++) {
    float inv = 1.f / l_i[r];
    const size_t rb = base + (size_t)(qrow + r + 8 * lh) * DM;
#pragma unroll
    for (int f = 0; f < 4; f++)
      ob[rb + f * 16 + lr] = f2bf(o[f][r] * inv);
  }
}

// ---------------------------------------------------------------- launcher
extern "C" void kernel_launch(void* const* d_in, const int* in_sizes, int n_in,
                              void* d_out, int out_size, void* d_ws,
                              size_t ws_size, hipStream_t stream) {
  const float* x         = (const float*)d_in[0];
  const float* ln1_g     = (const float*)d_in[1];
  const float* ln1_b     = (const float*)d_in[2];
  const float* q_w       = (const float*)d_in[3];
  const float* k_w       = (const float*)d_in[4];
  const float* v_w       = (const float*)d_in[5];
  const float* o_w       = (const float*)d_in[6];
  const float* ln2_g     = (const float*)d_in[7];
  const float* ln2_b     = (const float*)d_in[8];
  const float* ffn_in_w  = (const float*)d_in[9];
  const float* ffn_in_b  = (const float*)d_in[10];
  const float* sw_w      = (const float*)d_in[11];
  const float* sw_v      = (const float*)d_in[12];
  const float* ffn_out_w = (const float*)d_in[13];
  const float* ffn_out_b = (const float*)d_in[14];
  float* out = (float*)d_out;

  const int D = 1024, F = 4096, NT = 4096;   // tokens = B*L
  char* ws = (char*)d_ws;
  const size_t MB = 1ull << 20;
  bf16_t* wq   = (bf16_t*)(ws + 0 * MB);     // 2MB each q/k/v/o
  bf16_t* wk   = (bf16_t*)(ws + 2 * MB);
  bf16_t* wv   = (bf16_t*)(ws + 4 * MB);
  bf16_t* wo   = (bf16_t*)(ws + 6 * MB);
  bf16_t* wfi  = (bf16_t*)(ws + 8 * MB);     // 8MB
  bf16_t* wsw  = (bf16_t*)(ws + 16 * MB);    // 32MB
  bf16_t* wsv  = (bf16_t*)(ws + 48 * MB);    // 32MB
  bf16_t* wfo  = (bf16_t*)(ws + 80 * MB);    // 8MB
  bf16_t* hb   = (bf16_t*)(ws + 88 * MB);    // 8MB (LN out, reused for LN2)
  bf16_t* qbuf = (bf16_t*)(ws + 96 * MB);    // 8MB
  bf16_t* kbuf = (bf16_t*)(ws + 104 * MB);   // 8MB
  bf16_t* vbuf = (bf16_t*)(ws + 112 * MB);   // 8MB
  bf16_t* aob  = (bf16_t*)(ws + 120 * MB);   // 8MB attn out
  float*  x1   = (float*) (ws + 128 * MB);   // 16MB residual stream
  bf16_t* ub   = (bf16_t*)(ws + 96 * MB);    // 32MB, reuses q/k/v/aob region
  float*  abuf = (float*) (ws + 144 * MB);   // 64MB  u@sw_w^T
  bf16_t* gb   = (bf16_t*)(ws + 208 * MB);   // 32MB  SwiGLU out

  dim3 blk(256);
  // 1) weights -> bf16
  cast_f32_to_bf16<<<2048, blk, 0, stream>>>(q_w, wq, D * D);
  cast_f32_to_bf16<<<2048, blk, 0, stream>>>(k_w, wk, D * D);
  cast_f32_to_bf16<<<2048, blk, 0, stream>>>(v_w, wv, D * D);
  cast_f32_to_bf16<<<2048, blk, 0, stream>>>(o_w, wo, D * D);
  cast_f32_to_bf16<<<2048, blk, 0, stream>>>(ffn_in_w, wfi, F * D);
  cast_f32_to_bf16<<<4096, blk, 0, stream>>>(sw_w, wsw, F * F);
  cast_f32_to_bf16<<<4096, blk, 0, stream>>>(sw_v, wsv, F * F);
  cast_f32_to_bf16<<<2048, blk, 0, stream>>>(ffn_out_w, wfo, D * F);

  // 2) LN1
  layernorm_bf16<<<NT / 8, blk, 0, stream>>>(x, ln1_g, ln1_b, hb, NT);

  // 3) Q/K/V projections (bf16 out)
  gemm_wmma_bf16<1><<<dim3(D / 128, NT / 128), blk, 0, stream>>>(
      hb, wq, nullptr, nullptr, nullptr, nullptr, qbuf, NT, D, D);
  gemm_wmma_bf16<1><<<dim3(D / 128, NT / 128), blk, 0, stream>>>(
      hb, wk, nullptr, nullptr, nullptr, nullptr, kbuf, NT, D, D);
  gemm_wmma_bf16<1><<<dim3(D / 128, NT / 128), blk, 0, stream>>>(
      hb, wv, nullptr, nullptr, nullptr, nullptr, vbuf, NT, D, D);

  // 4) flash attention
  flash_attn_wmma<<<dim3(2048 / 64, 16, 2), dim3(128), 0, stream>>>(
      qbuf, kbuf, vbuf, aob);

  // 5) x1 = x + attn @ o_w^T  (f32 out + residual)
  gemm_wmma_bf16<0><<<dim3(D / 128, NT / 128), blk, 0, stream>>>(
      aob, wo, nullptr, x, nullptr, x1, nullptr, NT, D, D);

  // 6) LN2
  layernorm_bf16<<<NT / 8, blk, 0, stream>>>(x1, ln2_g, ln2_b, hb, NT);

  // 7) u = h @ ffn_in^T + b  (bf16 out)
  gemm_wmma_bf16<1><<<dim3(F / 128, NT / 128), blk, 0, stream>>>(
      hb, wfi, ffn_in_b, nullptr, nullptr, nullptr, ub, NT, F, D);

  // 8) a = u @ sw_w^T  (f32 out)
  gemm_wmma_bf16<0><<<dim3(F / 128, NT / 128), blk, 0, stream>>>(
      ub, wsw, nullptr, nullptr, nullptr, abuf, nullptr, NT, F, F);

  // 9) g = silu(a) * (u @ sw_v^T)  (bf16 out, fused SwiGLU epilogue)
  gemm_wmma_bf16<2><<<dim3(F / 128, NT / 128), blk, 0, stream>>>(
      ub, wsv, nullptr, nullptr, abuf, nullptr, gb, NT, F, F);

  // 10) out = x1 + g @ ffn_out^T + b  (f32 to d_out)
  gemm_wmma_bf16<0><<<dim3(D / 128, NT / 128), blk, 0, stream>>>(
      gb, wfo, ffn_out_b, x1, nullptr, out, nullptr, NT, D, F);
}
